// VGGTCrossFrameCFAngleLoss_3350074491453
// MI455X (gfx1250) — matrix operand order repaired and checked by reading, not compiled, not run on hardware
//
#include <hip/hip_runtime.h>
#include <stdint.h>

// ---- problem constants (from reference) ----
#define BB 4
#define VV 8
#define PP 1369
#define DD 2048
#define RR 128
#define FF 3
#define KK 4
#define EE (4*PP)        // 5476 extra rows per batch
#define EP 5488          // 343*16 padded sim stride
#define ETILES 343
#define MM 384           // F*S
#define NBROWS 768       // c(512) + ref_t(128) + ref_s(128)
#define NCOLS 640        // ss(512) + sr(128)
#define EPSF 1e-8f
#define NTOT 786432      // B*F*R*S*K = 4*3*128*128*4

typedef __attribute__((ext_vector_type(16))) __bf16 v16bf;
typedef __attribute__((ext_vector_type(8)))  float  v8f;
typedef __attribute__((ext_vector_type(4)))  float  v4f;
typedef __attribute__((ext_vector_type(4)))  __bf16 v4bf;

union FragU { uint4 u[2]; v16bf v; };

__device__ __forceinline__ unsigned short bfbits(float x) {
    __bf16 h = (__bf16)x;                       // v_cvt (RNE)
    return __builtin_bit_cast(unsigned short, h);
}

// ---------------- K0: zero scalar output ----------------
__global__ void k_zero(float* out) { out[0] = 0.0f; }

// ---------------- K1: gather shared/ref rows -> bf16, row sum-of-squares ----
// grid.x = B*1024 rows; block = 256
__global__ void k_gather(const float* __restrict__ teacher,
                         const float* __restrict__ student,
                         const int* __restrict__ ref_perm,
                         const int* __restrict__ shared_perm,
                         unsigned short* __restrict__ Abuf,   // [B][2][384][D] bf16
                         unsigned short* __restrict__ Bmat,   // [B][768][D] bf16
                         float* __restrict__ rr,              // [B][2][128]
                         float* __restrict__ rnorm,           // [B][128]
                         float* __restrict__ hh)              // [B][2][384]
{
    __shared__ float red[256];
    int row = blockIdx.x;
    int b = row >> 10;
    int j = row & 1023;
    const float* src;
    unsigned short* dst;
    float* sq_out;
    float* sq_out2 = nullptr;
    if (j < 384) {                        // shared_t: frames 2,4,6
        int f = j / 128, s = j % 128;
        src = teacher + ((size_t)((b*VV + 2*(f+1)))*PP + shared_perm[s])*DD;
        dst = Abuf + ((size_t)((b*2+0)*MM + j))*DD;
        sq_out = hh + (b*2+0)*MM + j;
    } else if (j < 768) {                 // shared_s: student idx 1,2,3
        int j2 = j - 384; int f = j2 / 128, s = j2 % 128;
        src = student + ((size_t)((b*4 + (f+1)))*PP + shared_perm[s])*DD;
        dst = Abuf + ((size_t)((b*2+1)*MM + j2))*DD;
        sq_out = hh + (b*2+1)*MM + j2;
    } else if (j < 896) {                 // ref_t
        int r = j - 768;
        src = teacher + ((size_t)(b*VV + 0)*PP + ref_perm[r])*DD;
        dst = Bmat + ((size_t)(b*NBROWS + 512 + r))*DD;
        sq_out = rr + (b*2+0)*RR + r;
        sq_out2 = rnorm + b*RR + r;
    } else {                              // ref_s
        int r = j - 896;
        src = student + ((size_t)(b*4 + 0)*PP + ref_perm[r])*DD;
        dst = Bmat + ((size_t)(b*NBROWS + 640 + r))*DD;
        sq_out = rr + (b*2+1)*RR + r;
    }
    int t = threadIdx.x;
    float acc = 0.0f;
    for (int q = 0; q < DD/256; ++q) {
        int i = t + 256*q;
        float x = src[i];
        acc += x*x;
        dst[i] = bfbits(x);
    }
    red[t] = acc; __syncthreads();
    for (int s = 128; s > 0; s >>= 1) { if (t < s) red[t] += red[t+s]; __syncthreads(); }
    if (t == 0) { *sq_out = red[0]; if (sq_out2) *sq_out2 = sqrtf(red[0]); }
}

// ---------------- K2: sim GEMM (WMMA bf16), fused extra-row norm ------------
// grid = (343, B); block 256 (8 waves). Each wave: one 16(R)x16(E) tile; block
// covers R=128 for a 16-wide E tile. Extra frames streamed once (NT loads),
// converted to bf16 in ping-pong LDS tiles; one barrier per K=64 chunk.
__global__ void k_sim(const float* __restrict__ teacher,
                      const unsigned short* __restrict__ Bmat,  // ref_t rows at 512..639
                      const float* __restrict__ rnorm,
                      float* __restrict__ sim)                  // [B][128][EP]
{
    __shared__ __align__(16) __bf16 tile[2][16*72];   // [buf][erow*72], 64 valid/row
    __shared__ float sqpart[16][16];
    __shared__ float enorm[16];
    int b = blockIdx.y;
    int ebase = blockIdx.x * 16;
    int t = threadIdx.x;
    int lane = t & 31, wave = t >> 5;
    int hl = lane >> 4, l16 = lane & 15;

    // staging assignment (fixed across K chunks so sumsq accumulates per row)
    int srow = t >> 4;                // 0..15 e-row in tile
    int scol = (t & 15) * 4;          // 0..60
    int e = ebase + srow; if (e >= EE) e = EE - 1;
    int frame = 1 + 2*(e / PP);
    int p = e % PP;
    const float* esrc = teacher + ((size_t)(b*VV + frame)*PP + p)*DD;

    const unsigned short* arow = Bmat + ((size_t)(b*NBROWS + 512 + 16*wave + l16))*DD;

    v8f acc = {};
    float sq = 0.0f;

    // prime chunk 0
    {
        v4f xv = __builtin_nontemporal_load((const v4f*)(esrc + scol));
        sq += xv.x*xv.x + xv.y*xv.y + xv.z*xv.z + xv.w*xv.w;
        v4bf c4 = __builtin_convertvector(xv, v4bf);
        *(uint2*)&tile[0][srow*72 + scol] = __builtin_bit_cast(uint2, c4);
    }
    __syncthreads();

    for (int c = 0; c < DD/64; ++c) {
        int kbase = c*64;
        int cur = c & 1, nxt = cur ^ 1;
        v4f nx = {};
        bool have = (c + 1 < DD/64);
        if (have) nx = __builtin_nontemporal_load((const v4f*)(esrc + kbase + 64 + scol));
        #pragma unroll
        for (int sc = 0; sc < 2; ++sc) {
            int k0 = kbase + 32*sc;
            int ko = 32*sc;
            FragU a, bf;
            a.u[0]  = *(const uint4*)(arow + k0 + 16*hl);
            a.u[1]  = *(const uint4*)(arow + k0 + 16*hl + 8);
            bf.u[0] = *(const uint4*)(&tile[cur][l16*72 + ko + 16*hl]);
            bf.u[1] = *(const uint4*)(&tile[cur][l16*72 + ko + 16*hl + 8]);
            acc = __builtin_amdgcn_wmma_f32_16x16x32_bf16(false, a.v, false, bf.v,
                                                          (short)0, acc, false, false);
        }
        if (have) {
            sq += nx.x*nx.x + nx.y*nx.y + nx.z*nx.z + nx.w*nx.w;
            v4bf c4 = __builtin_convertvector(nx, v4bf);
            *(uint2*)&tile[nxt][srow*72 + scol] = __builtin_bit_cast(uint2, c4);
        }
        __syncthreads();   // nxt staged + cur reads retired before reuse
    }

    sqpart[srow][t & 15] = sq;
    __syncthreads();
    if (t < 16) {
        float s = 0.0f;
        #pragma unroll
        for (int i = 0; i < 16; ++i) s += sqpart[t][i];
        enorm[t] = sqrtf(s);
    }
    __syncthreads();
    int ecol = ebase + l16;
    if (ecol < EE) {
        float en = fmaxf(enorm[l16], 1e-12f);
        #pragma unroll
        for (int v = 0; v < 8; ++v) {
            int r = 16*wave + v + 8*hl;
            float rn = fmaxf(rnorm[b*RR + r], 1e-12f);
            sim[((size_t)(b*RR + r))*EP + ecol] = acc[v] / (rn * en);
        }
    }
}

// ---------------- K3: top-4 per (b,r) row ----------------
// grid = B*R blocks of 32 (one wave)
__global__ void k_topk(const float* __restrict__ sim, int* __restrict__ idx)
{
    __shared__ float vals[128];
    __shared__ int   inds[128];
    int row = blockIdx.x;                    // b*128 + r
    const float* srow = sim + (size_t)row*EP;
    int lane = threadIdx.x;
    float v0=-1e30f,v1=-1e30f,v2=-1e30f,v3=-1e30f;
    int   i0=-1,i1=-1,i2=-1,i3=-1;
    for (int e = lane; e < EE; e += 32) {
        float v = srow[e];
        if (v > v3) {
            if (v > v0)      { v3=v2;i3=i2; v2=v1;i2=i1; v1=v0;i1=i0; v0=v;i0=e; }
            else if (v > v1) { v3=v2;i3=i2; v2=v1;i2=i1; v1=v;i1=e; }
            else if (v > v2) { v3=v2;i3=i2; v2=v;i2=e; }
            else             { v3=v;i3=e; }
        }
    }
    vals[lane*4+0]=v0; inds[lane*4+0]=i0;
    vals[lane*4+1]=v1; inds[lane*4+1]=i1;
    vals[lane*4+2]=v2; inds[lane*4+2]=i2;
    vals[lane*4+3]=v3; inds[lane*4+3]=i3;
    __syncthreads();
    if (lane == 0) {
        for (int k = 0; k < 4; ++k) {
            float bv = -1e30f; int bi = 0x7fffffff; int bj = 0;
            for (int j = 0; j < 128; ++j) {
                float v = vals[j]; int i = inds[j];
                if (v > bv || (v == bv && i < bi)) { bv = v; bi = i; bj = j; }
            }
            idx[row*4 + k] = bi;
            vals[bj] = -1e30f;
        }
    }
}

// ---------------- K4: gather c rows -> bf16 + rk/kk dots (f32) --------------
// grid = B*R*K blocks of 256
__global__ void k_gatherc(const float* __restrict__ teacher,
                          const float* __restrict__ student,
                          const int* __restrict__ ref_perm,
                          const int* __restrict__ idx,
                          unsigned short* __restrict__ Bmat,
                          float* __restrict__ kk,      // [B][R][K]
                          float* __restrict__ rk)      // [B][2][R][K]
{
    __shared__ float red[256*3];
    int g = blockIdx.x;                 // b*512 + r*4 + k
    int b = g >> 9;
    int rk4 = g & 511;
    int r = rk4 >> 2;
    int e = idx[g];
    int frame = 1 + 2*(e / PP);
    int p = e % PP;
    const float* csrc = teacher + ((size_t)(b*VV + frame)*PP + p)*DD;
    const float* rt   = teacher + ((size_t)(b*VV + 0)*PP + ref_perm[r])*DD;
    const float* rs   = student + ((size_t)(b*4 + 0)*PP + ref_perm[r])*DD;
    unsigned short* dst = Bmat + ((size_t)(b*NBROWS + rk4))*DD;
    int t = threadIdx.x;
    float akk = 0.0f, art = 0.0f, ars = 0.0f;
    for (int q = 0; q < DD/256; ++q) {
        int i = t + 256*q;
        float c = csrc[i];
        akk += c*c;
        art += c * rt[i];
        ars += c * rs[i];
        dst[i] = bfbits(c);
    }
    red[t] = akk; red[256+t] = art; red[512+t] = ars;
    __syncthreads();
    for (int s = 128; s > 0; s >>= 1) {
        if (t < s) { red[t]+=red[t+s]; red[256+t]+=red[256+t+s]; red[512+t]+=red[512+t+s]; }
        __syncthreads();
    }
    if (t == 0) {
        kk[g] = red[0];
        rk[(b*2+0)*512 + rk4] = red[256];
        rk[(b*2+1)*512 + rk4] = red[512];
    }
}

// ---------------- K5: ss/sr GEMM (WMMA bf16) ----------------
// Cout[b][side][m][n]: n<512 -> ss (n=r*4+k), n>=512 -> sr (n=512+r).
// grid = 960 blocks of 256 (8 waves); one 16x16 tile per wave; 7680 tiles.
// Operands are L2-resident (~25 MB bf16) -> direct global fragment loads.
__global__ void k_gemm(const unsigned short* __restrict__ Abuf,
                       const unsigned short* __restrict__ Bmat,
                       float* __restrict__ Cout)
{
    int t = threadIdx.x;
    int lane = t & 31, wave = t >> 5;
    int g = blockIdx.x * 8 + wave;
    int bs = g / 960;                  // b*2 + side
    int tt = g % 960;
    int mtile = tt / 40;
    int ntile = tt % 40;
    int b = bs >> 1, side = bs & 1;
    int hl = lane >> 4, l16 = lane & 15;

    int n = ntile*16 + l16;                                // output col 0..639
    int brow = (side == 0) ? n : (n < 512 ? n : n + 128);  // ref_s rows at 640..767
    const unsigned short* arow = Abuf + ((size_t)(bs*MM + mtile*16 + l16))*DD;
    const unsigned short* brp  = Bmat + ((size_t)(b*NBROWS + brow))*DD;
    v8f acc = {};
    for (int c = 0; c < DD/64; ++c) {
        int kbase = c*64;
        #pragma unroll
        for (int sc = 0; sc < 2; ++sc) {
            int k0 = kbase + 32*sc;
            FragU a, bf;
            a.u[0]  = *(const uint4*)(arow + k0 + 16*hl);
            a.u[1]  = *(const uint4*)(arow + k0 + 16*hl + 8);
            bf.u[0] = *(const uint4*)(brp + k0 + 16*hl);
            bf.u[1] = *(const uint4*)(brp + k0 + 16*hl + 8);
            acc = __builtin_amdgcn_wmma_f32_16x16x32_bf16(false, a.v, false, bf.v,
                                                          (short)0, acc, false, false);
        }
    }
    #pragma unroll
    for (int v = 0; v < 8; ++v) {
        int m = mtile*16 + v + 8*hl;
        Cout[((size_t)(bs*MM + m))*NCOLS + ntile*16 + l16] = acc[v];
    }
}

// ---------------- K6: angles + Huber + reduction ----------------
// grid = NTOT/256 blocks of 256
__global__ void k_loss(const float* __restrict__ Cout,
                       const float* __restrict__ rr,
                       const float* __restrict__ hh,
                       const float* __restrict__ rk,
                       const float* __restrict__ kk,
                       float* __restrict__ out)
{
    __shared__ float red[256];
    int id = blockIdx.x*256 + threadIdx.x;   // b*196608 + f*65536 + r*512 + s*4 + k
    int k = id & 3;
    int s = (id >> 2) & 127;
    int r = (id >> 9) & 127;
    int fb = id >> 16;                       // = b*3 + f
    int f = fb % 3;
    int b = fb / 3;
    int m = f*128 + s;

    size_t rowT = ((size_t)((b*2+0)*MM + m))*NCOLS;
    size_t rowS = ((size_t)((b*2+1)*MM + m))*NCOLS;
    float ss_t = Cout[rowT + r*4 + k];
    float sr_t = Cout[rowT + 512 + r];
    float ss_s = Cout[rowS + r*4 + k];
    float sr_s = Cout[rowS + 512 + r];
    float kkv  = kk[b*512 + r*4 + k];
    float rk_t = rk[(b*2+0)*512 + r*4 + k];
    float rk_s = rk[(b*2+1)*512 + r*4 + k];
    float rr_t = rr[(b*2+0)*RR + r];
    float rr_s = rr[(b*2+1)*RR + r];
    float hh_t = hh[(b*2+0)*MM + m];
    float hh_s = hh[(b*2+1)*MM + m];

    auto angles = [](float ss, float sr, float rkv, float rrv, float kv, float hv,
                     float& a1, float& a2, float& a3) {
        float n_hr = sqrtf(fmaxf(hv + rrv - 2.0f*sr,  0.0f));
        float n_cr = sqrtf(fmaxf(kv + rrv - 2.0f*rkv, 0.0f));
        float n_hc = sqrtf(fmaxf(hv + kv  - 2.0f*ss,  0.0f));
        a1 = (ss - sr - rkv + rrv) / (fmaxf(n_hr, EPSF) * fmaxf(n_cr, EPSF));
        a2 = (sr - rkv - ss + kv)  / (fmaxf(n_cr, EPSF) * fmaxf(n_hc, EPSF));
        a3 = (rkv - sr - ss + hv)  / (fmaxf(n_hr, EPSF) * fmaxf(n_hc, EPSF));
    };
    float a1t,a2t,a3t,a1s,a2s,a3s;
    angles(ss_t, sr_t, rk_t, rr_t, kkv, hh_t, a1t,a2t,a3t);
    angles(ss_s, sr_s, rk_s, rr_s, kkv, hh_s, a1s,a2s,a3s);
    auto hub = [](float p, float q) {
        float e = p - q; float ae = fabsf(e);
        return ae <= 1.0f ? 0.5f*e*e : (ae - 0.5f);
    };
    float val = hub(a1s,a1t) + hub(a2s,a2t) + hub(a3s,a3t);

    red[threadIdx.x] = val; __syncthreads();
    for (int st = 128; st > 0; st >>= 1) {
        if (threadIdx.x < st) red[threadIdx.x] += red[threadIdx.x+st];
        __syncthreads();
    }
    if (threadIdx.x == 0) atomicAdd(out, red[0] * (1.0f/(float)NTOT));
}

// ---------------- host: pipeline launch ----------------
extern "C" void kernel_launch(void* const* d_in, const int* in_sizes, int n_in,
                              void* d_out, int out_size, void* d_ws, size_t ws_size,
                              hipStream_t stream) {
    (void)in_sizes; (void)n_in; (void)out_size; (void)ws_size;
    const float* teacher = (const float*)d_in[0];
    const float* student = (const float*)d_in[1];
    const int* ref_perm    = (const int*)d_in[2];
    const int* shared_perm = (const int*)d_in[3];
    float* out = (float*)d_out;

    char* ws = (char*)d_ws;
    size_t off = 0;
    auto take = [&](size_t bytes) { size_t r = off; off += (bytes + 255) & ~(size_t)255; return r; };
    unsigned short* Abuf = (unsigned short*)(ws + take((size_t)BB*2*MM*DD*2));      // 12.6 MB
    unsigned short* Bmat = (unsigned short*)(ws + take((size_t)BB*NBROWS*DD*2));    // 12.6 MB
    float* sim   = (float*)(ws + take((size_t)BB*RR*EP*4));                         // 11.2 MB
    float* Cout  = (float*)(ws + take((size_t)BB*2*MM*NCOLS*4));                    //  7.9 MB
    int*   idxb  = (int*)  (ws + take((size_t)BB*RR*KK*4));
    float* rr    = (float*)(ws + take((size_t)BB*2*RR*4));
    float* rnorm = (float*)(ws + take((size_t)BB*RR*4));
    float* hh    = (float*)(ws + take((size_t)BB*2*MM*4));
    float* rk    = (float*)(ws + take((size_t)BB*2*RR*KK*4));
    float* kk    = (float*)(ws + take((size_t)BB*RR*KK*4));

    k_zero<<<1, 1, 0, stream>>>(out);
    k_gather<<<BB*1024, 256, 0, stream>>>(teacher, student, ref_perm, shared_perm,
                                          Abuf, Bmat, rr, rnorm, hh);
    k_sim<<<dim3(ETILES, BB), 256, 0, stream>>>(teacher, Bmat, rnorm, sim);
    k_topk<<<BB*RR, 32, 0, stream>>>(sim, idxb);
    k_gatherc<<<BB*RR*KK, 256, 0, stream>>>(teacher, student, ref_perm, idxb,
                                            Bmat, kk, rk);
    k_gemm<<<960, 256, 0, stream>>>(Abuf, Bmat, Cout);
    k_loss<<<NTOT/256, 256, 0, stream>>>(Cout, rr, hh, rk, kk, out);
}
